// Swin_attn_16131897164208
// MI455X (gfx1250) — matrix operand order repaired and checked by reading, not compiled, not run on hardware
//
#include <hip/hip_runtime.h>

typedef _Float16 v4h  __attribute__((ext_vector_type(4)));
typedef _Float16 v8h  __attribute__((ext_vector_type(8)));
typedef _Float16 v16h __attribute__((ext_vector_type(16)));
typedef float    v8f  __attribute__((ext_vector_type(8)));
typedef unsigned int u32x4 __attribute__((ext_vector_type(4)));
typedef int          i32x8 __attribute__((ext_vector_type(8)));
typedef int          i32x4 __attribute__((ext_vector_type(4)));

#define DIM   256
#define NHEAD 8
#define HD    32
#define NTOK  64

// LDS strides (elements), padded to avoid bank conflicts on strided b128 loads
#define XS_LD 264   // 256 + 8 halves  (row stride 528B -> +4 banks/row)
#define VT_LD 72    // 64 + 8 halves
#define SC_LD 68    // 64 + 4 floats
#define PB_LD 72    // 64 + 8 halves

// LDS layout (bytes); regions reused across barrier-separated phases
#define OFF_XS 0
#define SZ_XS  (NTOK*XS_LD*2)            // 33792
#define OFF_Q  (OFF_XS + SZ_XS)          // 33792
#define OFF_K  (OFF_Q + SZ_XS)           // 67584
#define OFF_VT (OFF_K + SZ_XS)           // 101376
#define SZ_VT  (DIM*VT_LD*2)             // 36864
#define OFF_SC (OFF_VT + SZ_VT)          // 138240
#define SZ_SC  (NHEAD*NTOK*SC_LD*4)      // 139264
#define OFF_PB 0                         // probs overlay xs/q/k (dead after scores)
#define OFF_AO OFF_SC                    // attn-out overlays scores (dead after softmax)
#define OFF_STAGE OFF_SC                 // f32 x staging (64KB) overlays scores (pre-phase-4)
#define SMEM_TOTAL (OFF_SC + SZ_SC)      // 277504 <= 320KB

// ---- WMMA operand loaders (contiguous-k order used by this file, wave32) ----
// A 16x32: lane<16 holds K {0..15}, lane>=16 holds K {16..31}
__device__ __forceinline__ v16h ld_a16(const _Float16* p) {
  v8h a = *(const v8h*)p;
  v8h b = *(const v8h*)(p + 8);
  return __builtin_shufflevector(a, b, 0,1,2,3,4,5,6,7,8,9,10,11,12,13,14,15);
}
// B 32x16: lane holds 16 contiguous K values of one column
__device__ __forceinline__ v16h ld_b16(const _Float16* p) {
  v8h a = *(const v8h*)p;
  v8h b = *(const v8h*)(p + 8);
  return __builtin_shufflevector(a, b, 0,1,2,3,4,5,6,7,8,9,10,11,12,13,14,15);
}
__device__ __forceinline__ v8f wmma16(v16h a, v16h b, v8f c) {
  return __builtin_amdgcn_wmma_f32_16x16x32_f16(false, a, false, b, (short)0, c,
                                                false, false);
}

// ---- weight f32 -> f16 pre-pass ----
__global__ void cvt_weights(const float* __restrict__ qkv_w,
                            const float* __restrict__ proj_w,
                            _Float16* __restrict__ wq, _Float16* __restrict__ wp) {
  int i = blockIdx.x * blockDim.x + threadIdx.x;
  if (i < 3*DIM*DIM) wq[i] = (_Float16)qkv_w[i];
  if (i < DIM*DIM)   wp[i] = (_Float16)proj_w[i];
}

__global__ __launch_bounds__(256, 1)
void swin_window_attn(const float* __restrict__ x,
                      const _Float16* __restrict__ wq,
                      const float* __restrict__ qkv_b,
                      const _Float16* __restrict__ wp,
                      const float* __restrict__ proj_b,
                      float* __restrict__ out)
{
  extern __shared__ char smem[];
  _Float16* xs = (_Float16*)(smem + OFF_XS);
  _Float16* qs = (_Float16*)(smem + OFF_Q);
  _Float16* ks = (_Float16*)(smem + OFF_K);
  _Float16* vt = (_Float16*)(smem + OFF_VT);   // v transposed: [channel][token]
  float*    sc = (float*)   (smem + OFF_SC);
  _Float16* pb = (_Float16*)(smem + OFF_PB);
  _Float16* ao = (_Float16*)(smem + OFF_AO);
  const float* stage = (const float*)(smem + OFF_STAGE); // TDM-staged f32 x tile

  const int tid  = threadIdx.x;
  const int wave = tid >> 5;
  const int lane = tid & 31;
  const int l16  = lane & 15;
  const int hi   = lane >> 4;

  const int win  = blockIdx.x;          // 2048 windows
  const int b    = win >> 6;
  const int winh = (win >> 3) & 7;
  const int winw = win & 7;

  // ---- Phase 0: TDM async load of the x window into LDS (f32 staging) ----
  // 2D tile: 8 rows (gh) x 2048 contiguous f32 (8 tokens * 256 ch),
  // row stride = W*C = 16384 elements. Issued once per block by wave 0.
  if (wave == 0) {
    unsigned long long gaddr =
        (unsigned long long)(x + (((size_t)(b*64 + winh*8)*64 + winw*8) * DIM));
    u32x4 g0;
    g0.x = 1u;                                      // count=1, user mode
    g0.y = (unsigned)OFF_STAGE;                     // lds_addr (bytes)
    g0.z = (unsigned)(gaddr & 0xffffffffu);         // global_addr[31:0]
    g0.w = (unsigned)((gaddr >> 32) & 0x01ffffffu)  // global_addr[56:32]
         | (2u << 30);                              // type=2 ("image")
    i32x8 g1;
    g1[0] = (2 << 16);                 // workgroup_mask=0, data_size=2 (4 bytes)
    g1[1] = (int)(16384u << 16);       // tensor_dim0 = 16384 (low 16 @ bits63:48)
    g1[2] = (8 << 16);                 // tensor_dim0 hi=0 | tensor_dim1=8
    g1[3] = (int)(2048u << 16);        // tensor_dim1 hi=0 | tile_dim0=2048
    g1[4] = 8;                         // tile_dim1=8, tile_dim2=0
    g1[5] = 16384;                     // tensor_dim0_stride (low 32)
    g1[6] = 0;                         // stride hi, tensor_dim1_stride lo
    g1[7] = 0;
    i32x4 z4 = {0, 0, 0, 0};
    i32x8 z8 = {0, 0, 0, 0, 0, 0, 0, 0};
    __builtin_amdgcn_tensor_load_to_lds(g0, g1, z4, z4, z8, 0);
    __builtin_amdgcn_s_wait_tensorcnt(0);
  }
  // hint the weight stream into cache while TDM runs
  __builtin_prefetch(wq + (wave*96 + l16)*DIM, 0, 3);
  __syncthreads();

  // ---- Phase 1: LDS f32 staging -> LDS f16 x tile ----
  for (int i = tid; i < NTOK * (DIM/4); i += 256) {
    int t  = i >> 6;
    int c4 = (i & 63) << 2;
    float4 f = *(const float4*)(stage + t*DIM + c4);
    v4h h; h.x = (_Float16)f.x; h.y = (_Float16)f.y;
           h.z = (_Float16)f.z; h.w = (_Float16)f.w;
    *(v4h*)(xs + t*XS_LD + c4) = h;
  }
  __syncthreads();

  // ---- Phase 2: QKV GEMM (64x256)@(256x768) -> q/k/vT in LDS ----
  for (int ti = 0; ti < 6; ++ti) {
    int nt = wave*6 + ti;               // 48 N-tiles over 8 waves
    // region select is tile-uniform (a 16-wide tile never straddles 256):
    // force it scalar so the store path is an s_cbranch, not exec-mask ops.
    int region = __builtin_amdgcn_readfirstlane(nt >> 4);  // 0=q, 1=k, 2=v
    int c  = nt*16 + l16;               // output channel 0..767
    float bias = qkv_b[c];
    for (int mt = 0; mt < 4; ++mt) {
      v8f acc = {};
      const _Float16* arow = xs + (mt*16 + l16)*XS_LD + hi*16;
      const _Float16* brow = wq + c*DIM + hi*16;
      #pragma unroll
      for (int kb = 0; kb < 8; ++kb)
        acc = wmma16(ld_a16(arow + kb*32), ld_b16(brow + kb*32), acc);
      int ch = c & 255;
      if (region == 0) {
        #pragma unroll
        for (int j = 0; j < 8; ++j)
          qs[(mt*16 + j + hi*8)*XS_LD + ch] = (_Float16)(acc[j] + bias);
      } else if (region == 1) {
        #pragma unroll
        for (int j = 0; j < 8; ++j)
          ks[(mt*16 + j + hi*8)*XS_LD + ch] = (_Float16)(acc[j] + bias);
      } else {
        #pragma unroll
        for (int j = 0; j < 8; ++j)
          vt[ch*VT_LD + (mt*16 + j + hi*8)] = (_Float16)(acc[j] + bias);
      }
    }
  }
  __syncthreads();

  // ---- Phase 3: 2D axial RoPE on q (with 1/sqrt(hd) scale) and k ----
  for (int m = 0; m < 2; ++m) {
    _Float16* mat = m ? ks : qs;
    float sm = m ? 1.0f : 0.17677669529663687f;   // hd^-0.5 folded into q
    for (int idx = tid; idx < NTOK*NHEAD*16; idx += 256) {
      int t    = idx >> 7;
      int rem  = idx & 127;
      int head = rem >> 4;
      int pr   = rem & 15;
      int p, pos, d;
      if (pr < 8) { p = pr;     pos = t >> 3; d = 2*pr; }          // row axis
      else        { p = pr - 8; pos = t & 7;  d = 16 + 2*(pr-8); } // col axis
      float ang = (float)pos * __expf(-1.1512925464970229f * (float)p); // 10000^(-p/8)
      float cs, sn; __sincosf(ang, &sn, &cs);
      int c0 = head*HD + d;
      float x1 = (float)mat[t*XS_LD + c0];
      float x2 = (float)mat[t*XS_LD + c0 + 1];
      mat[t*XS_LD + c0]     = (_Float16)(sm * (x1*cs - x2*sn));
      mat[t*XS_LD + c0 + 1] = (_Float16)(sm * (x1*sn + x2*cs));
    }
  }
  __syncthreads();

  // ---- Phase 4: scores = q @ k^T, one head per wave (64x64, K=32) ----
  {
    const int h = wave;
    float* shead = sc + h*NTOK*SC_LD;
    for (int mt = 0; mt < 4; ++mt) {
      const _Float16* arow = qs + (mt*16 + l16)*XS_LD + h*HD + hi*16;
      for (int nt = 0; nt < 4; ++nt) {
        const _Float16* brow = ks + (nt*16 + l16)*XS_LD + h*HD + hi*16;
        v8f acc = {};
        acc = wmma16(ld_a16(arow), ld_b16(brow), acc);
        #pragma unroll
        for (int j = 0; j < 8; ++j)
          shead[(mt*16 + j + hi*8)*SC_LD + nt*16 + l16] = acc[j];
      }
    }
  }
  __syncthreads();   // probs region overlays q/k/xs -> must be globally dead

  // ---- Phase 5: row softmax (f32) -> probs f16, vectorized LDS traffic ----
  {
    const int h = wave;
    float* shead = sc + h*NTOK*SC_LD;
    _Float16* phead = pb + h*NTOK*PB_LD;
    for (int r = lane; r < NTOK; r += 32) {
      const float4* row4 = (const float4*)(shead + r*SC_LD);
      float mx = -1e30f;
      #pragma unroll
      for (int cc = 0; cc < 16; ++cc) {
        float4 v = row4[cc];
        mx = fmaxf(mx, fmaxf(fmaxf(v.x, v.y), fmaxf(v.z, v.w)));
      }
      float sum = 0.f;
      v4h* prow4 = (v4h*)(phead + r*PB_LD);
      #pragma unroll
      for (int cc = 0; cc < 16; ++cc) {
        float4 v = row4[cc];
        float e0 = __expf(v.x - mx), e1 = __expf(v.y - mx);
        float e2 = __expf(v.z - mx), e3 = __expf(v.w - mx);
        sum += (e0 + e1) + (e2 + e3);
        v4h hh; hh.x = (_Float16)e0; hh.y = (_Float16)e1;
                hh.z = (_Float16)e2; hh.w = (_Float16)e3;
        prow4[cc] = hh;
      }
      float rinv = 1.0f / sum;
      #pragma unroll
      for (int cc = 0; cc < 16; ++cc) {
        v4h hh = prow4[cc];
        hh.x = (_Float16)((float)hh.x * rinv);
        hh.y = (_Float16)((float)hh.y * rinv);
        hh.z = (_Float16)((float)hh.z * rinv);
        hh.w = (_Float16)((float)hh.w * rinv);
        prow4[cc] = hh;
      }
    }
  }
  __syncthreads();   // attn-out overlays scores -> all softmax reads must finish

  // ---- Phase 6: attn @ v  (64x32, K=64), vT gives contiguous B columns ----
  {
    const int h = wave;
    const _Float16* phead = pb + h*NTOK*PB_LD;
    for (int mt = 0; mt < 4; ++mt) {
      for (int nt = 0; nt < 2; ++nt) {
        v8f acc = {};
        #pragma unroll
        for (int kb = 0; kb < 2; ++kb) {
          v16h a  = ld_a16(phead + (mt*16 + l16)*PB_LD + kb*32 + hi*16);
          v16h bm = ld_b16(vt + (h*HD + nt*16 + l16)*VT_LD + kb*32 + hi*16);
          acc = wmma16(a, bm, acc);
        }
        int c = h*HD + nt*16 + l16;
        #pragma unroll
        for (int j = 0; j < 8; ++j)
          ao[(mt*16 + j + hi*8)*XS_LD + c] = (_Float16)acc[j];
      }
    }
  }
  __syncthreads();

  // ---- Phase 7: projection (64x256)@(256x256)+b, scatter to (B,H*W,C) ----
  for (int ti = 0; ti < 2; ++ti) {
    int nt = wave*2 + ti;               // 16 N-tiles over 8 waves
    int c  = nt*16 + l16;
    float bias = proj_b[c];
    for (int mt = 0; mt < 4; ++mt) {
      v8f acc = {};
      const _Float16* arow = ao + (mt*16 + l16)*XS_LD + hi*16;
      const _Float16* brow = wp + c*DIM + hi*16;
      #pragma unroll
      for (int kb = 0; kb < 8; ++kb)
        acc = wmma16(ld_a16(arow + kb*32), ld_b16(brow + kb*32), acc);
      #pragma unroll
      for (int j = 0; j < 8; ++j) {
        int t  = mt*16 + j + hi*8;
        int gh = winh*8 + (t >> 3);
        int gw = winw*8 + (t & 7);
        out[((size_t)b*4096 + gh*64 + gw)*DIM + c] = acc[j] + bias;
      }
    }
  }
}

extern "C" void kernel_launch(void* const* d_in, const int* in_sizes, int n_in,
                              void* d_out, int out_size, void* d_ws, size_t ws_size,
                              hipStream_t stream) {
  const float* x      = (const float*)d_in[0];
  const float* qkv_w  = (const float*)d_in[1];
  const float* qkv_b  = (const float*)d_in[2];
  const float* proj_w = (const float*)d_in[3];
  const float* proj_b = (const float*)d_in[4];
  float* out = (float*)d_out;

  _Float16* wq = (_Float16*)d_ws;            // 3*256*256 f16 = 384KB
  _Float16* wp = wq + 3*DIM*DIM;             // 256*256 f16 = 128KB

  cvt_weights<<<768, 256, 0, stream>>>(qkv_w, proj_w, wq, wp);
  swin_window_attn<<<2048, 256, SMEM_TOTAL, stream>>>(x, wq, qkv_b, wp, proj_b, out);
}
